// PINNConvDecoder_37847251812839
// MI455X (gfx1250) — matrix operand order repaired and checked by reading, not compiled, not run on hardware
//
#include <hip/hip_runtime.h>
#include <cmath>

// ---------------------------------------------------------------------------
// PINN conv decoder for MI455X (gfx1250, wave32, WMMA).
// 3-layer MLP done as LDS-resident GEMMs with v_wmma_f32_16x16x32_f16.
// ---------------------------------------------------------------------------

typedef __attribute__((ext_vector_type(16))) _Float16 v16h;
typedef __attribute__((ext_vector_type(8)))  _Float16 v8h;
typedef __attribute__((ext_vector_type(8)))  float    v8f;

constexpr int   NPTS   = 8192;     // points per batch element
constexpr int   CCH    = 64;       // z_grid channels
constexpr int   GH     = 64, GW = 64;
constexpr int   HID    = 128;
constexpr int   NFREQ  = 64;
constexpr int   DIN    = CCH + 2 * NFREQ + 2;   // 194
constexpr int   DINP   = 224;                   // padded to 7*32
constexpr int   TILE_M = 128;                   // points per block
constexpr int   SA     = 232;  // activation row stride in halves (bank-friendly)
constexpr int   SW     = 136;  // weight row stride in halves

constexpr float TEMPORAL_MAX = 3.0f;
constexpr float MIN_STD      = 0.01f;
constexpr float TWO_PI       = 6.28318530717958647692f;

// LDS carve (bytes, all 16B aligned)
constexpr int OFF_ACT0 = 0;
constexpr int OFF_ACT1 = OFF_ACT0 + TILE_M * SA * 2;   // 59392
constexpr int OFF_W    = OFF_ACT1 + TILE_M * SA * 2;   // 118784
constexpr int OFF_HW   = OFF_W + DINP * SW * 2;        // 179712
constexpr int OFF_HB   = OFF_HW + HID * 8 * 4;         // 183808
constexpr int SMEM_BYTES = OFF_HB + 16;                // 183824

// A fragment: 16x32 f16, M = lane&15, K chunks per ISA 7.12.2 table.
__device__ __forceinline__ v16h load_afrag(const _Float16* base, int row16, int kc) {
    const int lane = threadIdx.x & 31;
    const _Float16* p = base + (row16 + (lane & 15)) * SA + kc * 32 + ((lane >> 4) << 3);
    v8h lo = *(const v8h*)p;          // K = kb .. kb+7
    v8h hi = *(const v8h*)(p + 16);   // K = kb+16 .. kb+23
    return __builtin_shufflevector(lo, hi, 0,1,2,3,4,5,6,7,8,9,10,11,12,13,14,15);
}

// B fragment: 32x16 f16, N = lane&15, K in the same table order as the A fragment.
__device__ __forceinline__ v16h load_bfrag(const _Float16* wbase, int col16, int kc) {
    const int lane = threadIdx.x & 31;
    const _Float16* p = wbase + (kc * 32 + ((lane >> 4) << 3)) * SW + col16 + (lane & 15);
    v16h b;
#pragma unroll
    for (int s = 0; s < 8; ++s) { b[s] = p[s * SW]; b[s + 8] = p[(s + 16) * SW]; }
    return b;
}

// Stage a KxHID fp32 weight matrix into LDS as f16, zero-padded to Kpad rows.
__device__ __forceinline__ void load_weights(const float* __restrict__ W,
                                             _Float16* sWgt, int K, int Kpad) {
    for (int i = threadIdx.x; i < Kpad * HID; i += 256) {
        const int k = i >> 7, n = i & 127;
        const float v = (k < K) ? W[k * HID + n] : 0.0f;
        sWgt[k * SW + n] = (_Float16)v;
    }
}

// One MLP layer: out = tanh(in @ W + b), 128x(KC*32)x128 via WMMA.
template <int KC>
__device__ __forceinline__ void mlp_layer(const _Float16* sIn, const _Float16* sWgt,
                                          const float* __restrict__ bias,
                                          _Float16* sOut) {
    const int wave = threadIdx.x >> 5;
    const int lane = threadIdx.x & 31;
    const int rt   = wave * 16;          // this wave's 16-row strip

    v8f acc[8] = {};
#pragma unroll
    for (int kc = 0; kc < KC; ++kc) {
        const v16h a = load_afrag(sIn, rt, kc);
#pragma unroll
        for (int ct = 0; ct < 8; ++ct) {
            const v16h b = load_bfrag(sWgt, ct * 16, kc);
            acc[ct] = __builtin_amdgcn_wmma_f32_16x16x32_f16(
                false, a, false, b, (short)0, acc[ct], false, false);
        }
    }
    // D layout: VGPR j -> M = j + 8*(lane>>4); N = lane&15
    const int n0 = lane & 15;
    const int mh = (lane >> 4) << 3;
#pragma unroll
    for (int ct = 0; ct < 8; ++ct) {
        const float bb = bias[ct * 16 + n0];
#pragma unroll
        for (int j = 0; j < 8; ++j) {
            sOut[(rt + mh + j) * SA + ct * 16 + n0] = (_Float16)tanhf(acc[ct][j] + bb);
        }
    }
}

__global__ __launch_bounds__(256)
void pinn_decoder_wmma(const float* __restrict__ coords,
                       const float* __restrict__ z_grid,
                       const float* __restrict__ fourierB,
                       const float* __restrict__ W1, const float* __restrict__ b1,
                       const float* __restrict__ W2, const float* __restrict__ b2,
                       const float* __restrict__ W3, const float* __restrict__ b3,
                       const float* __restrict__ Wvel, const float* __restrict__ bvel,
                       const float* __restrict__ Wfh,  const float* __restrict__ bfh,
                       const float* __restrict__ Wph,  const float* __restrict__ bph,
                       const float* __restrict__ Wmu,  const float* __restrict__ bmu,
                       const float* __restrict__ Wstd, const float* __restrict__ bstd,
                       float* __restrict__ out) {
    extern __shared__ char smem[];
    _Float16* act0 = (_Float16*)(smem + OFF_ACT0);
    _Float16* act1 = (_Float16*)(smem + OFF_ACT1);
    _Float16* sW   = (_Float16*)(smem + OFF_W);
    float*    sHW  = (float*)(smem + OFF_HW);
    float*    sHB  = (float*)(smem + OFF_HB);

    const int tid   = threadIdx.x;
    const int pbase = blockIdx.x * TILE_M;            // global point base
    const int bidx  = pbase / NPTS;                   // batch element
    const float* zg = z_grid + (size_t)bidx * CCH * GH * GW;

    // ---- stage 0: featurize 128 points into act0 (two threads per point) ----
    {
        const int p    = tid & 127;
        const int half = tid >> 7;
        const int gp   = pbase + p;
        const float x = coords[gp * 3 + 0];
        const float y = coords[gp * 3 + 1];
        const float t = coords[gp * 3 + 2];

        const float gx = (x + 1.0f) * 0.5f * (GW - 1);
        const float gy = (y + 1.0f) * 0.5f * (GH - 1);
        const float x0f = floorf(gx), y0f = floorf(gy);
        const int x0 = min(max((int)x0f, 0), GW - 1);
        const int x1 = min(max((int)x0f + 1, 0), GW - 1);
        const int y0 = min(max((int)y0f, 0), GH - 1);
        const int y1 = min(max((int)y0f + 1, 0), GH - 1);
        const float fx = gx - x0f, fy = gy - y0f;
        const float wa = (1.0f - fx) * (1.0f - fy);
        const float wb = (1.0f - fx) * fy;
        const float wc = fx * (1.0f - fy);
        const float wd = fx * fy;

        _Float16* row = act0 + p * SA;
        for (int c = half; c < CCH; c += 2) {
            const float* g = zg + c * (GH * GW);
            const float v = wa * g[y0 * GW + x0] + wb * g[y1 * GW + x0]
                          + wc * g[y0 * GW + x1] + wd * g[y1 * GW + x1];
            row[c] = (_Float16)v;
        }
        const float tp = TWO_PI * (t / TEMPORAL_MAX);
        for (int f = half; f < NFREQ; f += 2) {
            const float pr = tp * fourierB[f];
            row[CCH + f]          = (_Float16)sinf(pr);
            row[CCH + NFREQ + f]  = (_Float16)cosf(pr);
        }
        if (half == 0) {
            row[192] = (_Float16)x;
            row[193] = (_Float16)y;
#pragma unroll
            for (int kk = DIN; kk < DINP; ++kk) row[kk] = (_Float16)0.0f;
        }
    }

    load_weights(W1, sW, DIN, DINP);

    // head weight matrix (128x8) + biases into LDS, fp32
    for (int i = tid; i < HID * 8; i += 256) {
        const int k = i >> 3, o = i & 7;
        float v;
        if      (o < 2)  v = Wvel[k * 2 + o];
        else if (o < 4)  v = Wfh[k * 2 + (o - 2)];
        else if (o < 6)  v = Wph[k * 2 + (o - 4)];
        else if (o == 6) v = Wmu[k];
        else             v = Wstd[k];
        sHW[i] = v;
    }
    if (tid < 8) {
        float v;
        if      (tid < 2)  v = bvel[tid];
        else if (tid < 4)  v = bfh[tid - 2];
        else if (tid < 6)  v = bph[tid - 4];
        else if (tid == 6) v = bmu[0];
        else               v = bstd[0];
        sHB[tid] = v;
    }
    __syncthreads();

    // ---- 3 WMMA layers, ping-ponging LDS activation buffers ----
    mlp_layer<DINP / 32>(act0, sW, b1, act1);
    __syncthreads();                       // act1 done; sW free
    load_weights(W2, sW, HID, HID);
    __syncthreads();
    mlp_layer<HID / 32>(act1, sW, b2, act0);
    __syncthreads();
    load_weights(W3, sW, HID, HID);
    __syncthreads();
    mlp_layer<HID / 32>(act0, sW, b3, act1);   // act1 = feat
    __syncthreads();

    // ---- heads: 128x8 GEMV on VALU, coalesced fp32 output ----
    {
        const int p   = tid >> 1;
        const int grp = tid & 1;                 // outputs [grp*4 .. grp*4+3]
        const _Float16* feat = act1 + p * SA;
        float a0 = 0.f, a1 = 0.f, a2 = 0.f, a3 = 0.f;
#pragma unroll 4
        for (int k = 0; k < HID; ++k) {
            const float f = (float)feat[k];
            const float* w = sHW + k * 8 + grp * 4;
            a0 += f * w[0]; a1 += f * w[1]; a2 += f * w[2]; a3 += f * w[3];
        }
        float r0 = a0 + sHB[grp * 4 + 0];
        float r1 = a1 + sHB[grp * 4 + 1];
        float r2 = a2 + sHB[grp * 4 + 2];
        float r3 = a3 + sHB[grp * 4 + 3];
        if (grp == 1) {  // col 7 = s_std: softplus + MIN_STD
            r3 = (r3 > 20.0f) ? r3 : log1pf(expf(r3));
            r3 += MIN_STD;
        }
        float* o = out + (size_t)(pbase + p) * 8 + grp * 4;
        o[0] = r0; o[1] = r1; o[2] = r2; o[3] = r3;
    }
}

extern "C" void kernel_launch(void* const* d_in, const int* in_sizes, int n_in,
                              void* d_out, int out_size, void* d_ws, size_t ws_size,
                              hipStream_t stream) {
    (void)in_sizes; (void)n_in; (void)d_ws; (void)ws_size; (void)out_size;

    const float* coords   = (const float*)d_in[0];
    const float* z_grid   = (const float*)d_in[1];
    const float* fourierB = (const float*)d_in[2];
    const float* W1   = (const float*)d_in[3];
    const float* b1   = (const float*)d_in[4];
    const float* W2   = (const float*)d_in[5];
    const float* b2   = (const float*)d_in[6];
    const float* W3   = (const float*)d_in[7];
    const float* b3   = (const float*)d_in[8];
    const float* Wvel = (const float*)d_in[9];
    const float* bvel = (const float*)d_in[10];
    const float* Wfh  = (const float*)d_in[11];
    const float* bfh  = (const float*)d_in[12];
    const float* Wph  = (const float*)d_in[13];
    const float* bph  = (const float*)d_in[14];
    const float* Wmu  = (const float*)d_in[15];
    const float* bmu  = (const float*)d_in[16];
    const float* Wstd = (const float*)d_in[17];
    const float* bstd = (const float*)d_in[18];
    float* out = (float*)d_out;

    // allow >64KB dynamic LDS (gfx1250 WGP has 320KB)
    static bool attr_done = false;
    if (!attr_done) {
        (void)hipFuncSetAttribute((const void*)pinn_decoder_wmma,
                                  hipFuncAttributeMaxDynamicSharedMemorySize,
                                  SMEM_BYTES);
        attr_done = true;
    }

    const int total_points = 16 * NPTS;          // B*N = 131072
    const int blocks = total_points / TILE_M;    // 1024
    pinn_decoder_wmma<<<blocks, 256, SMEM_BYTES, stream>>>(
        coords, z_grid, fourierB, W1, b1, W2, b2, W3, b3,
        Wvel, bvel, Wfh, bfh, Wph, bph, Wmu, bmu, Wstd, bstd, out);
}
